// MultiHeadLatentAttention_61753039782415
// MI455X (gfx1250) — matrix operand order repaired and checked by reading, not compiled, not run on hardware
//
#include <hip/hip_runtime.h>

// ---------------------------------------------------------------------------
// MLA forward for MI455X (gfx1250, wave32, WMMA + async LDS staging).
// All matmuls are A(MxK) @ Bt(NxK)^T with K contiguous in both operands,
// run through v_wmma_f32_16x16x32_bf16 (bf16 in, f32 accumulate).
// Attention K/V tiles are staged into LDS with GLOBAL_LOAD_ASYNC_TO_LDS_B128
// (ASYNCcnt), double-buffered across 64-key chunks.  Inner loops are ordered
// so that groups of fragment loads precede runs of INDEPENDENT WMMAs.
// ---------------------------------------------------------------------------

typedef __bf16 v16bf __attribute__((ext_vector_type(16)));
typedef float  v8f   __attribute__((ext_vector_type(8)));

#define B_DIM    2
#define S_DIM    2048
#define DMODEL   2048
#define H_DIM    16
#define DLAT     512
#define DHEAD    128
#define MROWS    (B_DIM * S_DIM)        // 4096 flattened (b, s) rows

__device__ __forceinline__ v8f zero8() {
  v8f z;
#pragma unroll
  for (int i = 0; i < 8; ++i) z[i] = 0.0f;
  return z;
}

// A-matrix fragment, 16x32 bf16 (MxK), contiguous K order (matches B frags):
// lanes 0-15 : M=lane,     elements = K{0..15}
// lanes 16-31: M=lane-16,  elements = K{16..31}
template <typename T>
__device__ __forceinline__ v16bf load_afrag(const T* __restrict__ tile, int ld, int lane) {
  const int r = lane & 15;
  const int g = lane >> 4;
  const T* q = tile + (size_t)r * ld + g * 16;
  v16bf f;
#pragma unroll
  for (int i = 0; i < 8; ++i) f[i]     = (__bf16)q[i];
#pragma unroll
  for (int i = 0; i < 8; ++i) f[i + 8] = (__bf16)q[8 + i];
  return f;
}

// B-matrix fragment, 32x16 bf16 (KxN): lanes 0-15: N=lane, K=0..15 sequential;
// lanes 16-31: N=lane-16, K=16..31.  Source is Bt (N rows x K cols, row-major).
template <typename T>
__device__ __forceinline__ v16bf load_bfrag(const T* __restrict__ tile, int ld, int lane) {
  const int r = lane & 15;
  const int g = lane >> 4;
  const T* q = tile + (size_t)r * ld + g * 16;
  v16bf f;
#pragma unroll
  for (int i = 0; i < 16; ++i) f[i] = (__bf16)q[i];
  return f;
}

__device__ __forceinline__ v8f wmma_bf16(v16bf a, v16bf b, v8f c) {
  // (neg_a, A, neg_b, B, c_mod, C, reuse_a, reuse_b)
  return __builtin_amdgcn_wmma_f32_16x16x32_bf16(false, a, false, b, (short)0, c, false, false);
}

// Async copy of 16 bytes global -> LDS (per lane), tracked by ASYNCcnt.
__device__ __forceinline__ void async16(unsigned lds_addr, const void* gptr) {
  asm volatile("global_load_async_to_lds_b128 %0, %1, off"
               :: "v"(lds_addr), "v"(gptr)
               : "memory");
}

__device__ __forceinline__ void wait_async0() {
  asm volatile("s_wait_asynccnt 0x0" ::: "memory");
}

// ---------------------------------------------------------------------------
// C = A(MxK) @ Bt(NxK)^T.  Block tile 128x128, 4 waves of 64x64, K step 32.
// OUT_BF16: store bf16 (intermediates) else f32.  TRANS: store C^T (for Vt).
// ---------------------------------------------------------------------------
template <typename AT, typename BT, bool OUT_BF16, bool TRANS>
__global__ void __launch_bounds__(128)
gemm_abt(const AT* __restrict__ A, const BT* __restrict__ Bt, void* __restrict__ Cout,
         int M, int N, int K, int lda, int ldb, int ldc) {
  const int lane = threadIdx.x & 31;
  const int wave = threadIdx.x >> 5;
  const int bm = blockIdx.y * 128 + (wave >> 1) * 64;
  const int bn = blockIdx.x * 128 + (wave & 1) * 64;
  const int r = lane & 15, g = lane >> 4;
  (void)M; (void)N;

  v8f acc[4][4];
#pragma unroll
  for (int i = 0; i < 4; ++i)
#pragma unroll
    for (int j = 0; j < 4; ++j) acc[i][j] = zero8();

  for (int kb = 0; kb < K; kb += 32) {
    v16bf af[4];
#pragma unroll
    for (int i = 0; i < 4; ++i)
      af[i] = load_afrag<AT>(A + (size_t)(bm + i * 16) * lda + kb, lda, lane);
    // Batch B-fragment loads in pairs, then 8 independent WMMAs.
#pragma unroll
    for (int jj = 0; jj < 4; jj += 2) {
      v16bf bf0 = load_bfrag<BT>(Bt + (size_t)(bn + jj * 16) * ldb + kb, ldb, lane);
      v16bf bf1 = load_bfrag<BT>(Bt + (size_t)(bn + (jj + 1) * 16) * ldb + kb, ldb, lane);
#pragma unroll
      for (int i = 0; i < 4; ++i) acc[i][jj]     = wmma_bf16(af[i], bf0, acc[i][jj]);
#pragma unroll
      for (int i = 0; i < 4; ++i) acc[i][jj + 1] = wmma_bf16(af[i], bf1, acc[i][jj + 1]);
    }
  }

  // D layout: lane r = N within tile, element e -> M = 8*g + e.
#pragma unroll
  for (int i = 0; i < 4; ++i)
#pragma unroll
    for (int j = 0; j < 4; ++j)
#pragma unroll
      for (int e = 0; e < 8; ++e) {
        const int m = bm + i * 16 + 8 * g + e;
        const int n = bn + j * 16 + r;
        const float v = acc[i][j][e];
        if (OUT_BF16) {
          __bf16* C = (__bf16*)Cout;
          if (TRANS) C[(size_t)n * ldc + m] = (__bf16)v;
          else       C[(size_t)m * ldc + n] = (__bf16)v;
        } else {
          float* C = (float*)Cout;
          C[(size_t)m * ldc + n] = v;
        }
      }
}

// ---------------------------------------------------------------------------
// Flash-style attention.  Block = 4 waves; wave handles 16 query rows of one
// (b, h); streams keys in chunks of 64 with online softmax.
// Q,K: bf16 [(b*S+s)][(h*128+d)] ld=2048.  Vt: bf16 [(h*128+d)][(b*S+t)] ld=4096.
// K/V chunks are shared by all 4 waves -> staged once per block into LDS with
// double-buffered async copies.
// ---------------------------------------------------------------------------
__global__ void __launch_bounds__(128)
mla_attn(const __bf16* __restrict__ Qb, const __bf16* __restrict__ Kb,
         const __bf16* __restrict__ Vt, __bf16* __restrict__ Ob) {
  __shared__ __bf16 Kl[2][64][128];   // keys t (64) x d (128), 16 KB per buffer
  __shared__ __bf16 Vl[2][128][64];   // d (128) x keys t (64), 16 KB per buffer
  __shared__ __bf16 Pl[4][16][64];    // per-wave P tile (D-layout -> A-layout)

  const int tid  = threadIdx.x;
  const int lane = tid & 31;
  const int wave = tid >> 5;
  const int r = lane & 15, g = lane >> 4;
  const int b = blockIdx.z, h = blockIdx.y;
  const int m0 = blockIdx.x * 64 + wave * 16;

  const __bf16* qt = Qb + (size_t)(b * S_DIM + m0) * DMODEL + h * DHEAD;
  const __bf16* kt = Kb + (size_t)(b * S_DIM) * DMODEL + h * DHEAD;
  const __bf16* vt = Vt + (size_t)(h * DHEAD) * MROWS + b * S_DIM;

  v16bf qf[4];
#pragma unroll
  for (int c = 0; c < 4; ++c) qf[c] = load_afrag<__bf16>(qt + c * 32, DMODEL, lane);

  float mrow[8], lrow[8];
  v8f oacc[8];
#pragma unroll
  for (int e = 0; e < 8; ++e) { mrow[e] = -1.0e30f; lrow[e] = 0.0f; }
#pragma unroll
  for (int d = 0; d < 8; ++d) oacc[d] = zero8();

  const float scale = 0.08838834764831845f;   // 1/sqrt(128)

  // Stage one 64-key chunk (K: 16 KB, V: 16 KB) = 1024 x 16B chunks each,
  // spread over 128 threads x 8 copies.
  auto stage_kv = [&](int t0, int buf) {
    const unsigned kbase = (unsigned)(size_t)&Kl[buf][0][0];
    const unsigned vbase = (unsigned)(size_t)&Vl[buf][0][0];
#pragma unroll
    for (int c = 0; c < 8; ++c) {
      const int q = tid + c * 128;              // 0..1023
      const int trow = q >> 4, col = (q & 15) * 8;   // 16 chunks per K row
      async16(kbase + q * 16, kt + (size_t)(t0 + trow) * DMODEL + col);
    }
#pragma unroll
    for (int c = 0; c < 8; ++c) {
      const int q = tid + c * 128;
      const int drow = q >> 3, tcol = (q & 7) * 8;   // 8 chunks per V row
      async16(vbase + q * 16, vt + (size_t)drow * MROWS + t0 + tcol);
    }
  };

  stage_kv(0, 0);

  for (int it = 0; it < S_DIM / 64; ++it) {
    const int t0 = it * 64;
    const int buf = it & 1;

    wait_async0();        // this wave's copies for `buf` have landed
    __syncthreads();      // everyone's copies landed; prior reads of buf^1 done

    if (t0 + 64 < S_DIM) stage_kv(t0 + 64, buf ^ 1);   // prefetch next chunk

    // ---- scores: 16x64 = Q(16x128) @ K_chunk(64x128)^T, B-frags from LDS ----
    // c outer: load 4 K-fragments, then 4 INDEPENDENT WMMAs (distinct sacc[j]).
    v8f sacc[4];
#pragma unroll
    for (int j = 0; j < 4; ++j) sacc[j] = zero8();
#pragma unroll
    for (int c = 0; c < 4; ++c) {
      v16bf kf[4];
#pragma unroll
      for (int j = 0; j < 4; ++j)
        kf[j] = load_bfrag<__bf16>(&Kl[buf][j * 16][c * 32], 128, lane);
#pragma unroll
      for (int j = 0; j < 4; ++j)
        sacc[j] = wmma_bf16(qf[c], kf[j], sacc[j]);
    }

    // ---- online softmax over the 64-key chunk ----
    float mnew[8], corr[8], rsum[8];
#pragma unroll
    for (int e = 0; e < 8; ++e) {
      float v = fmaxf(fmaxf(sacc[0][e], sacc[1][e]), fmaxf(sacc[2][e], sacc[3][e])) * scale;
      v = fmaxf(v, __shfl_xor(v, 1, 32));
      v = fmaxf(v, __shfl_xor(v, 2, 32));
      v = fmaxf(v, __shfl_xor(v, 4, 32));
      v = fmaxf(v, __shfl_xor(v, 8, 32));   // reduce within 16-lane half (same 8 rows)
      mnew[e] = fmaxf(mrow[e], v);
      corr[e] = __expf(mrow[e] - mnew[e]);
      rsum[e] = 0.0f;
    }
#pragma unroll
    for (int j = 0; j < 4; ++j)
#pragma unroll
      for (int e = 0; e < 8; ++e) {
        const float p = __expf(sacc[j][e] * scale - mnew[e]);
        rsum[e] += p;
        Pl[wave][8 * g + e][j * 16 + r] = (__bf16)p;   // row-major P in LDS
      }
#pragma unroll
    for (int e = 0; e < 8; ++e) {
      float v = rsum[e];
      v += __shfl_xor(v, 1, 32);
      v += __shfl_xor(v, 2, 32);
      v += __shfl_xor(v, 4, 32);
      v += __shfl_xor(v, 8, 32);
      lrow[e] = lrow[e] * corr[e] + v;
      mrow[e] = mnew[e];
    }
#pragma unroll
    for (int d = 0; d < 8; ++d)
#pragma unroll
      for (int e = 0; e < 8; ++e) oacc[d][e] *= corr[e];

    // wave-private LDS region: only need this wave's stores visible to itself
    asm volatile("s_wait_dscnt 0" ::: "memory");

    // ---- O += P(16x64) @ V_chunk(64x128): Bt-frags = Vt rows from LDS ----
    // Per key-half: load 4 V-fragments, then 4 INDEPENDENT WMMAs (distinct oacc).
#pragma unroll
    for (int kc = 0; kc < 2; ++kc) {
      v16bf pf = load_afrag<__bf16>(&Pl[wave][0][0] + kc * 32, 64, lane);
#pragma unroll
      for (int dd = 0; dd < 8; dd += 4) {
        v16bf vf[4];
#pragma unroll
        for (int d = 0; d < 4; ++d)
          vf[d] = load_bfrag<__bf16>(&Vl[buf][(dd + d) * 16][kc * 32], 64, lane);
#pragma unroll
        for (int d = 0; d < 4; ++d)
          oacc[dd + d] = wmma_bf16(pf, vf[d], oacc[dd + d]);
      }
    }
  }

#pragma unroll
  for (int e = 0; e < 8; ++e) lrow[e] = 1.0f / lrow[e];
#pragma unroll
  for (int d = 0; d < 8; ++d)
#pragma unroll
    for (int e = 0; e < 8; ++e) {
      const int m = m0 + 8 * g + e;
      const int n = h * DHEAD + d * 16 + r;
      Ob[(size_t)(b * S_DIM + m) * DMODEL + n] = (__bf16)(oacc[d][e] * lrow[e]);
    }
}

// ---------------------------------------------------------------------------
extern "C" void kernel_launch(void* const* d_in, const int* in_sizes, int n_in,
                              void* d_out, int out_size, void* d_ws, size_t ws_size,
                              hipStream_t stream) {
  (void)in_sizes; (void)n_in; (void)out_size; (void)ws_size;

  const float* x    = (const float*)d_in[0];  // (B,S,DMODEL)
  const float* Wq   = (const float*)d_in[1];  // (DMODEL, DMODEL)
  const float* Wdkv = (const float*)d_in[2];  // (DLAT, DMODEL)
  const float* Wupk = (const float*)d_in[3];  // (H, DHEAD, DLAT) -> (DMODEL, DLAT)
  const float* Wupv = (const float*)d_in[4];  // (H, DHEAD, DLAT) -> (DMODEL, DLAT)
  const float* Wo   = (const float*)d_in[5];  // (DMODEL, DMODEL)
  float* out = (float*)d_out;

  // Workspace (bf16 intermediates, ~68 MiB total)
  __bf16* Qb = (__bf16*)d_ws;                         // [MROWS][DMODEL]
  __bf16* Ck = Qb + (size_t)MROWS * DMODEL;           // [MROWS][DLAT]
  __bf16* Kb = Ck + (size_t)MROWS * DLAT;             // [MROWS][DMODEL]
  __bf16* Vt = Kb + (size_t)MROWS * DMODEL;           // [DMODEL][MROWS] (transposed V)
  __bf16* Ob = Vt + (size_t)MROWS * DMODEL;           // [MROWS][DMODEL]

  const dim3 blk(128);

  // Q = x @ Wq^T
  gemm_abt<float, float, true, false>
      <<<dim3(DMODEL / 128, MROWS / 128), blk, 0, stream>>>(
          x, Wq, Qb, MROWS, DMODEL, DMODEL, DMODEL, DMODEL, DMODEL);

  // C_kv = x @ Wdkv^T
  gemm_abt<float, float, true, false>
      <<<dim3(DLAT / 128, MROWS / 128), blk, 0, stream>>>(
          x, Wdkv, Ck, MROWS, DLAT, DMODEL, DMODEL, DMODEL, DLAT);

  // K = C_kv @ Wupk^T   (columns are (h,d))
  gemm_abt<__bf16, float, true, false>
      <<<dim3(DMODEL / 128, MROWS / 128), blk, 0, stream>>>(
          Ck, Wupk, Kb, MROWS, DMODEL, DLAT, DLAT, DLAT, DMODEL);

  // V^T = (C_kv @ Wupv^T)^T   (stored [h*128+d][b*S+t], ld = MROWS)
  gemm_abt<__bf16, float, true, true>
      <<<dim3(DMODEL / 128, MROWS / 128), blk, 0, stream>>>(
          Ck, Wupv, Vt, MROWS, DMODEL, DLAT, DLAT, DLAT, MROWS);

  // softmax(Q K^T / sqrt(Dh)) @ V  -> Ob [(b,s)][(h,d)]
  mla_attn<<<dim3(S_DIM / 64, H_DIM, B_DIM), blk, 0, stream>>>(Qb, Kb, Vt, Ob);

  // out = Ob @ Wo^T  (f32 output)
  gemm_abt<__bf16, float, false, false>
      <<<dim3(DMODEL / 128, MROWS / 128), blk, 0, stream>>>(
          Ob, Wo, out, MROWS, DMODEL, DMODEL, DMODEL, DMODEL, DMODEL);
}
